// AttnBlock_32444182954762
// MI455X (gfx1250) — matrix operand order repaired and checked by reading, not compiled, not run on hardware
//
#include <hip/hip_runtime.h>
#include <stdint.h>

// ----------------------------- types -----------------------------
typedef __attribute__((ext_vector_type(16))) _Float16 v16h;
typedef __attribute__((ext_vector_type(8)))  _Float16 v8h;
typedef __attribute__((ext_vector_type(8)))  float    v8f;
typedef __attribute__((ext_vector_type(4)))  unsigned int u32x4;
typedef __attribute__((ext_vector_type(8)))  int      i32x8;
typedef __attribute__((ext_vector_type(4)))  int      i32x4;

// LDS (addrspace 3) pointer for guaranteed ds_load codegen
typedef const __attribute__((address_space(3))) v8h* lds_v8h_cp;

#define BATCH 8
#define CH    512
#define NPOS  4096
#define QSCALE 0.044194173824159216f   // 512^-0.5, folded into q

#define TILE_K        32
#define TILE_ROWS     128
#define LDS_ROW_BYTES 80               // 64B data + 16B TDM pad (16B aligned, conflict-free)
#define TILE_BYTES    (TILE_ROWS * LDS_ROW_BYTES)   // 10240
#define GEMM_LDS      (4 * TILE_BYTES)              // 40 KB: A/B double-buffered

union V16U { v16h v; v8h h[2]; };

// ---------------- TDM: async 2D tile load global -> LDS ----------------
// Loads a 128-row x 32-element (f16) tile from a row-major matrix with the
// given element stride. Descriptor per CDNA5 ISA.
// LDS padding: pad_interval=3 (16 DWORDs) + pad_amount=3 (4 DWORDs) -> 80B rows.
static __device__ __forceinline__ void tdm_load_tile(const void* g, int stride_elems,
                                                     uint32_t ldsoff)
{
  uint64_t ga = (uint64_t)(uintptr_t)g;
  u32x4 g0;
  g0[0] = 1u;                                              // count=1, user descriptor
  g0[1] = ldsoff;                                          // lds_addr (bytes)
  g0[2] = (uint32_t)ga;                                    // global_addr[31:0]
  g0[3] = (uint32_t)((ga >> 32) & 0x01FFFFFFu) | (2u << 30); // addr[56:32] | type=2
  i32x8 g1;
  uint32_t td0 = (uint32_t)stride_elems;                   // tensor_dim0 (= row stride)
  g1[0] = (int)((1u << 16)        // data_size = 1 (2 bytes)
              | (1u << 20)        // pad_enable
              | (3u << 22)        // pad_interval: 16 DWORDs
              | (3u << 25));      // pad_amount:   4 DWORDs
  g1[1] = (int)((td0 & 0xFFFFu) << 16);                    // tensor_dim0[15:0]
  g1[2] = (int)((td0 >> 16) | ((uint32_t)TILE_ROWS << 16));// tensor_dim0 hi | tensor_dim1 lo
  g1[3] = (int)((uint32_t)TILE_K << 16);                   // tensor_dim1 hi=0 | tile_dim0=32
  g1[4] = (int)TILE_ROWS;                                  // tile_dim1=128 | tile_dim2=0
  g1[5] = (int)td0;                                        // tensor_dim0_stride[31:0]
  g1[6] = 0;                                               // stride hi | dim1_stride lo
  g1[7] = 0;
  i32x4 z4 = {0, 0, 0, 0};
  i32x8 z8 = {0, 0, 0, 0, 0, 0, 0, 0};
  // clang-23 / therock-10.0 form: (g0, g1, g2, g3, g_extra, cpol)
  __builtin_amdgcn_tensor_load_to_lds(g0, g1, z4, z4, z8, 0);
}

// ---------------- WMMA fragment loads from padded LDS tiles ----------------
// Addresses are raw 32-bit LDS byte offsets -> ds_load_b128 codegen.
// A (16x32 f16, [m][k]): lane m=lane&15; half=lane>>4: K = {0..7,16..23}+8*half
static __device__ __forceinline__ v16h load_a_frag(uint32_t base, int row_base, int lane)
{
  const int m = lane & 15, half = lane >> 4;
  const uint32_t r = base + (uint32_t)(row_base + m) * LDS_ROW_BYTES + half * 16;
  V16U u;
  u.h[0] = *(lds_v8h_cp)(size_t)(r);
  u.h[1] = *(lds_v8h_cp)(size_t)(r + 32);
  return u.v;
}
// B (32x16 f16, [n][k] in LDS): lane n=lane&15; half=lane>>4: K = {0..7,16..23}+8*half
static __device__ __forceinline__ v16h load_b_frag(uint32_t base, int row_base, int lane)
{
  const int n = lane & 15, half = lane >> 4;
  const uint32_t r = base + (uint32_t)(row_base + n) * LDS_ROW_BYTES + half * 16;
  V16U u;
  u.h[0] = *(lds_v8h_cp)(size_t)(r);
  u.h[1] = *(lds_v8h_cp)(size_t)(r + 32);
  return u.v;
}

// ---------------- 128x128 GEMM core (8 waves, 64x32 per wave) ----------------
// A: [m][k] row-major f16 (tile rows m0..m0+127 passed in via Abase)
// B: [n][k] row-major f16 (tile rows n0..n0+127 passed in via Bbase)
// D[m][n] += sum_k A[m][k]*B[n][k]   (B operand is the k-major layout WMMA wants)
static __device__ __forceinline__ void gemm_tile_f16(const _Float16* __restrict__ Abase, int Astr,
                                                     const _Float16* __restrict__ Bbase, int Bstr,
                                                     int K, char* smem, v8f acc[4][2])
{
  const int tid  = threadIdx.x;
  const int lane = tid & 31;
  const int wid  = tid >> 5;
  const int wm   = (wid & 1) * 64;
  const int wn   = (wid >> 1) * 32;
  const uint32_t base = (uint32_t)(uintptr_t)smem;   // LDS byte offset of dynamic LDS
  const int KT = K / TILE_K;

  if (wid == 0) {   // one wave drives the TDM (EXEC-independent, per-wave op)
    tdm_load_tile(Abase, Astr, base);
    tdm_load_tile(Bbase, Bstr, base + 2 * TILE_BYTES);
  }
  for (int kt = 0; kt < KT; ++kt) {
    const uint32_t boff = (uint32_t)(kt & 1) * TILE_BYTES;
    if (wid == 0) {
      if (kt + 1 < KT) {
        const uint32_t noff = (uint32_t)((kt + 1) & 1) * TILE_BYTES;
        tdm_load_tile(Abase + (size_t)(kt + 1) * TILE_K, Astr, base + noff);
        tdm_load_tile(Bbase + (size_t)(kt + 1) * TILE_K, Bstr, base + 2 * TILE_BYTES + noff);
        __builtin_amdgcn_s_wait_tensorcnt(2);   // current pair landed, next pair in flight
      } else {
        __builtin_amdgcn_s_wait_tensorcnt(0);
      }
    }
    __syncthreads();
    const uint32_t la = base + boff;
    const uint32_t lb = base + 2 * TILE_BYTES + boff;
    v16h af[4], bf[2];
#pragma unroll
    for (int i = 0; i < 4; ++i) af[i] = load_a_frag(la, wm + i * 16, lane);
#pragma unroll
    for (int j = 0; j < 2; ++j) bf[j] = load_b_frag(lb, wn + j * 16, lane);
#pragma unroll
    for (int i = 0; i < 4; ++i)
#pragma unroll
      for (int j = 0; j < 2; ++j)
        acc[i][j] = __builtin_amdgcn_wmma_f32_16x16x32_f16(
            false, af[i], false, bf[j], (short)0, acc[i][j], false, false);
    __syncthreads();
  }
}

static __device__ __forceinline__ void zero_acc(v8f acc[4][2])
{
  const v8f z = {0.f, 0.f, 0.f, 0.f, 0.f, 0.f, 0.f, 0.f};
#pragma unroll
  for (int i = 0; i < 4; ++i)
#pragma unroll
    for (int j = 0; j < 2; ++j) acc[i][j] = z;
}

// ---------------- kernel 1: f32 weights -> f16 ----------------
__global__ void __launch_bounds__(256) wconv_kernel(const float* __restrict__ wq,
                                                    const float* __restrict__ wk,
                                                    const float* __restrict__ wv,
                                                    const float* __restrict__ wo,
                                                    _Float16* __restrict__ wh)
{
  const int i   = blockIdx.x * 256 + threadIdx.x;     // 4 * 512 * 512 total
  const int sel = i >> 18;
  const int r   = i & (CH * CH - 1);
  const float* src = (sel == 0) ? wq : (sel == 1) ? wk : (sel == 2) ? wv : wo;
  wh[i] = (_Float16)src[r];
}

// ---------------- kernel 2: GroupNorm -> hT [B][N][C] f16 ----------------
__global__ void __launch_bounds__(256) gn_kernel(const float* __restrict__ x,
                                                 const float* __restrict__ sc,
                                                 const float* __restrict__ bi,
                                                 _Float16* __restrict__ hT)
{
  const int tid = threadIdx.x;
  const int b   = blockIdx.x >> 5;
  const int g   = blockIdx.x & 31;
  const float* xp = x + ((size_t)b * CH + g * 16) * NPOS;

  float s = 0.f, ss = 0.f;
  for (int cc = 0; cc < 16; ++cc) {
    const float* row = xp + (size_t)cc * NPOS;
    for (int n = tid; n < NPOS; n += 256) { float v = row[n]; s += v; ss += v * v; }
  }
  __shared__ float rs[256], rq[256];
  rs[tid] = s; rq[tid] = ss;
  __syncthreads();
  for (int o = 128; o > 0; o >>= 1) {
    if (tid < o) { rs[tid] += rs[tid + o]; rq[tid] += rq[tid + o]; }
    __syncthreads();
  }
  const float mean = rs[0] * (1.f / 65536.f);
  const float var  = rq[0] * (1.f / 65536.f) - mean * mean;
  const float inv  = rsqrtf(var + 1e-6f);

  float scl[16], bsl[16];
#pragma unroll
  for (int cc = 0; cc < 16; ++cc) {
    const float sv = sc[g * 16 + cc] * inv;
    scl[cc] = sv;
    bsl[cc] = bi[g * 16 + cc] - mean * sv;
  }
  _Float16* hrow = hT + (size_t)b * NPOS * CH + g * 16;
  for (int n = tid; n < NPOS; n += 256) {
    v8h o0, o1;
#pragma unroll
    for (int cc = 0; cc < 8; ++cc)  o0[cc] = (_Float16)(xp[(size_t)cc * NPOS + n] * scl[cc] + bsl[cc]);
#pragma unroll
    for (int cc = 0; cc < 8; ++cc)  o1[cc] = (_Float16)(xp[(size_t)(cc + 8) * NPOS + n] * scl[cc + 8] + bsl[cc + 8]);
    _Float16* d = hrow + (size_t)n * CH;
    *(v8h*)(d)     = o0;
    *(v8h*)(d + 8) = o1;
  }
}

// ---------------- kernel 3: QKV GEMM ----------------
// D[o][i] = W[o][:] . hT[i][:]  ; q,k stored transposed [N][C], v natural [C][N]
__global__ void __launch_bounds__(256) qkv_kernel(const _Float16* __restrict__ wh,
                                                  const _Float16* __restrict__ hT,
                                                  const float* __restrict__ bq,
                                                  const float* __restrict__ bk,
                                                  const float* __restrict__ bv,
                                                  _Float16* __restrict__ qT,
                                                  _Float16* __restrict__ kT,
                                                  _Float16* __restrict__ vv)
{
  extern __shared__ char smem[];
  const int n0    = blockIdx.x * 128;           // i
  const int m0    = blockIdx.y * 128;           // o
  const int b     = blockIdx.z / 3;
  const int which = blockIdx.z % 3;

  v8f acc[4][2];
  zero_acc(acc);
  gemm_tile_f16(wh + (size_t)which * CH * CH + (size_t)m0 * CH, CH,
                hT + ((size_t)b * NPOS + n0) * CH, CH, CH, smem, acc);

  const int lane = threadIdx.x & 31, wid = threadIdx.x >> 5;
  const int wm = (wid & 1) * 64, wn = (wid >> 1) * 32;
  const int half = lane >> 4, ln = lane & 15;
  const float* bias = (which == 0) ? bq : (which == 1) ? bk : bv;

  if (which < 2) {
    _Float16* dst = (which == 0) ? qT : kT;
    const float sc = (which == 0) ? QSCALE : 1.0f;
#pragma unroll
    for (int mi = 0; mi < 4; ++mi)
#pragma unroll
      for (int ni = 0; ni < 2; ++ni) {
        const int i  = n0 + wn + ni * 16 + ln;
        const int ob = m0 + wm + mi * 16 + half * 8;
        v8h o;
#pragma unroll
        for (int r = 0; r < 8; ++r) o[r] = (_Float16)((acc[mi][ni][r] + bias[ob + r]) * sc);
        *(v8h*)(dst + ((size_t)b * NPOS + i) * CH + ob) = o;
      }
  } else {
#pragma unroll
    for (int mi = 0; mi < 4; ++mi)
#pragma unroll
      for (int ni = 0; ni < 2; ++ni) {
        const int i = n0 + wn + ni * 16 + ln;
#pragma unroll
        for (int r = 0; r < 8; ++r) {
          const int o = m0 + wm + mi * 16 + half * 8 + r;
          vv[((size_t)b * CH + o) * NPOS + i] = (_Float16)(acc[mi][ni][r] + bias[o]);
        }
      }
  }
}

// ---------------- kernel 4: S = qT . kT^T (one batch) ----------------
__global__ void __launch_bounds__(256) scores_kernel(const _Float16* __restrict__ qTb,
                                                     const _Float16* __restrict__ kTb,
                                                     float* __restrict__ S)
{
  extern __shared__ char smem[];
  const int n0 = blockIdx.x * 128;   // j
  const int m0 = blockIdx.y * 128;   // i
  v8f acc[4][2];
  zero_acc(acc);
  gemm_tile_f16(qTb + (size_t)m0 * CH, CH, kTb + (size_t)n0 * CH, CH, CH, smem, acc);

  const int lane = threadIdx.x & 31, wid = threadIdx.x >> 5;
  const int wm = (wid & 1) * 64, wn = (wid >> 1) * 32;
  const int half = lane >> 4, ln = lane & 15;
#pragma unroll
  for (int mi = 0; mi < 4; ++mi)
#pragma unroll
    for (int ni = 0; ni < 2; ++ni)
#pragma unroll
      for (int r = 0; r < 8; ++r) {
        const int i = m0 + wm + mi * 16 + half * 8 + r;
        const int j = n0 + wn + ni * 16 + ln;
        S[(size_t)i * NPOS + j] = acc[mi][ni][r];
      }
}

// ---------------- kernel 5: row softmax, S(f32) -> P(f16) ----------------
__global__ void __launch_bounds__(256) softmax_kernel(const float* __restrict__ S,
                                                      _Float16* __restrict__ P)
{
  const int tid = threadIdx.x;
  const float* row = S + (size_t)blockIdx.x * NPOS;
  float v[16];
  float mx = -3.4e38f;
#pragma unroll
  for (int t = 0; t < 16; ++t) { v[t] = row[tid + t * 256]; mx = fmaxf(mx, v[t]); }
  __shared__ float red[256];
  red[tid] = mx;
  __syncthreads();
  for (int o = 128; o > 0; o >>= 1) {
    if (tid < o) red[tid] = fmaxf(red[tid], red[tid + o]);
    __syncthreads();
  }
  const float m = red[0];
  __syncthreads();
  float s = 0.f;
#pragma unroll
  for (int t = 0; t < 16; ++t) { v[t] = __expf(v[t] - m); s += v[t]; }
  red[tid] = s;
  __syncthreads();
  for (int o = 128; o > 0; o >>= 1) {
    if (tid < o) red[tid] += red[tid + o];
    __syncthreads();
  }
  const float rs = 1.f / red[0];
  _Float16* prow = P + (size_t)blockIdx.x * NPOS;
#pragma unroll
  for (int t = 0; t < 16; ++t) prow[tid + t * 256] = (_Float16)(v[t] * rs);
}

// ---------------- kernel 6: O^T = P . v^T (one batch) ----------------
// OT[i][c] = sum_j P[i][j] * v[c][j]
__global__ void __launch_bounds__(256) av_kernel(const _Float16* __restrict__ P,
                                                 const _Float16* __restrict__ vb,
                                                 _Float16* __restrict__ oTb)
{
  extern __shared__ char smem[];
  const int n0 = blockIdx.x * 128;   // c
  const int m0 = blockIdx.y * 128;   // i
  v8f acc[4][2];
  zero_acc(acc);
  gemm_tile_f16(P + (size_t)m0 * NPOS, NPOS, vb + (size_t)n0 * NPOS, NPOS, NPOS, smem, acc);

  const int lane = threadIdx.x & 31, wid = threadIdx.x >> 5;
  const int wm = (wid & 1) * 64, wn = (wid >> 1) * 32;
  const int half = lane >> 4, ln = lane & 15;
#pragma unroll
  for (int mi = 0; mi < 4; ++mi)
#pragma unroll
    for (int ni = 0; ni < 2; ++ni)
#pragma unroll
      for (int r = 0; r < 8; ++r) {
        const int i = m0 + wm + mi * 16 + half * 8 + r;
        const int c = n0 + wn + ni * 16 + ln;
        oTb[(size_t)i * CH + c] = (_Float16)acc[mi][ni][r];
      }
}

// ---------------- kernel 7: out = x + wo . O + bo ----------------
__global__ void __launch_bounds__(256) proj_kernel(const _Float16* __restrict__ woh,
                                                   const _Float16* __restrict__ oT,
                                                   const float* __restrict__ x,
                                                   const float* __restrict__ bo,
                                                   float* __restrict__ out)
{
  extern __shared__ char smem[];
  const int n0 = blockIdx.x * 128;   // i
  const int m0 = blockIdx.y * 128;   // o
  const int b  = blockIdx.z;
  v8f acc[4][2];
  zero_acc(acc);
  gemm_tile_f16(woh + (size_t)m0 * CH, CH,
                oT + ((size_t)b * NPOS + n0) * CH, CH, CH, smem, acc);

  const int lane = threadIdx.x & 31, wid = threadIdx.x >> 5;
  const int wm = (wid & 1) * 64, wn = (wid >> 1) * 32;
  const int half = lane >> 4, ln = lane & 15;
#pragma unroll
  for (int mi = 0; mi < 4; ++mi)
#pragma unroll
    for (int ni = 0; ni < 2; ++ni)
#pragma unroll
      for (int r = 0; r < 8; ++r) {
        const int o = m0 + wm + mi * 16 + half * 8 + r;
        const int i = n0 + wn + ni * 16 + ln;
        const size_t idx = ((size_t)b * CH + o) * NPOS + i;
        out[idx] = x[idx] + bo[o] + acc[mi][ni][r];
      }
}

// ----------------------------- host launcher -----------------------------
extern "C" void kernel_launch(void* const* d_in, const int* in_sizes, int n_in,
                              void* d_out, int out_size, void* d_ws, size_t ws_size,
                              hipStream_t stream)
{
  (void)in_sizes; (void)n_in; (void)out_size; (void)ws_size;
  const float* x   = (const float*)d_in[0];
  const float* gsc = (const float*)d_in[1];
  const float* gbi = (const float*)d_in[2];
  const float* wq  = (const float*)d_in[3];
  const float* bq  = (const float*)d_in[4];
  const float* wk  = (const float*)d_in[5];
  const float* bk  = (const float*)d_in[6];
  const float* wv  = (const float*)d_in[7];
  const float* bv  = (const float*)d_in[8];
  const float* wo  = (const float*)d_in[9];
  const float* bo  = (const float*)d_in[10];
  float* out = (float*)d_out;

  // Workspace layout (258 MB): weights(2) | hT(32) | qT(32) | kT(32) | v(32) | OT(32) | S(64,f32) | P(32)
  const size_t MB = 1024 * 1024;
  char* ws = (char*)d_ws;
  _Float16* wh = (_Float16*)(ws);             // [4][512][512] f16 (q,k,v,o)
  _Float16* hT = (_Float16*)(ws + 2 * MB);    // [B][N][C]
  _Float16* qT = (_Float16*)(ws + 34 * MB);   // [B][N][C] (softmax scale folded)
  _Float16* kT = (_Float16*)(ws + 66 * MB);   // [B][N][C]
  _Float16* vv = (_Float16*)(ws + 98 * MB);   // [B][C][N]
  _Float16* oT = (_Float16*)(ws + 130 * MB);  // [B][N][C]
  float*    S  = (float*)   (ws + 162 * MB);  // [N][N] f32, reused per batch (L2-resident)
  _Float16* P  = (_Float16*)(ws + 226 * MB);  // [N][N] f16, reused per batch

  wconv_kernel<<<4096, 256, 0, stream>>>(wq, wk, wv, wo, wh);
  gn_kernel<<<BATCH * 32, 256, 0, stream>>>(x, gsc, gbi, hT);
  qkv_kernel<<<dim3(NPOS / 128, CH / 128, BATCH * 3), 256, GEMM_LDS, stream>>>(
      wh, hT, bq, bk, bv, qT, kT, vv);

  for (int b = 0; b < BATCH; ++b) {
    const _Float16* qTb = qT + (size_t)b * NPOS * CH;
    const _Float16* kTb = kT + (size_t)b * NPOS * CH;
    const _Float16* vb  = vv + (size_t)b * CH * NPOS;
    _Float16*       oTb = oT + (size_t)b * NPOS * CH;
    scores_kernel<<<dim3(NPOS / 128, NPOS / 128), 256, GEMM_LDS, stream>>>(qTb, kTb, S);
    softmax_kernel<<<NPOS, 256, 0, stream>>>(S, P);
    av_kernel<<<dim3(CH / 128, NPOS / 128), 256, GEMM_LDS, stream>>>(P, vb, oTb);
  }

  proj_kernel<<<dim3(NPOS / 128, CH / 128, BATCH), 256, GEMM_LDS, stream>>>(
      wh + (size_t)3 * CH * CH, oT, x, bo, out);
}
